// AdvantageModule_1434519077166
// MI455X (gfx1250) — matrix operand order repaired and checked by reading, not compiled, not run on hardware
//
#include <hip/hip_runtime.h>

#define BATCH    65536
#define KDIM     512
#define NTRIL    528
#define ADIM     32
#define MT       64           // batch rows per workgroup
#define NWAVES   4
#define NTHREADS 128
#define NTILES   33           // 528 / 16

typedef __attribute__((ext_vector_type(16))) __bf16 v16bf;
typedef __attribute__((ext_vector_type(8)))  __bf16 v8bf;
typedef __attribute__((ext_vector_type(4)))  __bf16 v4bf;
typedef __attribute__((ext_vector_type(8)))  float  v8f;

// LDS layout (bytes)
#define OFF_AHI     0
#define OFF_ALO     (MT * KDIM * 2)              // 65536
#define OFF_W       (OFF_ALO + MT * KDIM * 2)    // 131072: two W buffers follow
#define WTILE_BYTES (16 * KDIM * 2)              // 16384 (one hi or lo plane)
#define WBUF_STRIDE (2 * WTILE_BYTES)            // 32768 (hi+lo per buffer)
#define OFF_DIFF    (OFF_W + 2 * WBUF_STRIDE)    // 196608
#define OFF_Y       (OFF_DIFF + MT * ADIM * 4)   // 204800
#define LDS_BYTES   (OFF_Y + MT * ADIM * 4)      // 212992

// Compile-time tril index -> (row, col) lookup (row-major lower triangle).
struct TrilTab { unsigned char r[NTRIL]; unsigned char c[NTRIL]; };
constexpr TrilTab makeTrilTab() {
    TrilTab t{};
    int idx = 0;
    for (int rr = 0; rr < ADIM; ++rr)
        for (int cc = 0; cc <= rr; ++cc) {
            t.r[idx] = (unsigned char)rr;
            t.c[idx] = (unsigned char)cc;
            ++idx;
        }
    return t;
}
__constant__ TrilTab kTril = makeTrilTab();

__device__ __forceinline__ void split_bf16(float x, __bf16& hi, __bf16& lo) {
    __bf16 h = (__bf16)x;
    hi = h;
    lo = (__bf16)(x - (float)h);
}

// Generic pointer (known to point into LDS) -> 32-bit LDS byte address.
__device__ __forceinline__ unsigned lds_addr32(const void* p) {
    typedef __attribute__((address_space(3))) const char* lds_ptr;
    return (unsigned)(unsigned long long)(lds_ptr)p;
}

// Async DMA one W hi/lo tile: 16 rows x 512 bf16 each, contiguous.
__device__ __forceinline__ void async_w_tile(const __bf16* __restrict__ srcHi,
                                             const __bf16* __restrict__ srcLo,
                                             __bf16* dstHi, __bf16* dstLo, int t) {
    #pragma unroll
    for (int i = 0; i < (16 * KDIM) / (NTHREADS * 8); ++i) {   // 8 iters of 16B
        const int e = (i * NTHREADS + t) * 8;                  // element index
        const unsigned off = (unsigned)(e * 2);                // byte offset
        const unsigned ldsHi = lds_addr32(dstHi + e);
        const unsigned ldsLo = lds_addr32(dstLo + e);
        asm volatile("global_load_async_to_lds_b128 %0, %1, %2"
                     :: "v"(ldsHi), "v"(off), "s"(srcHi) : "memory");
        asm volatile("global_load_async_to_lds_b128 %0, %1, %2"
                     :: "v"(ldsLo), "v"(off), "s"(srcLo) : "memory");
    }
}

// Fallback: load f32 W tile and split to bf16 hi/lo in LDS.
__device__ __forceinline__ void convert_w_tile(const float* __restrict__ W, int n,
                                               __bf16* dstHi, __bf16* dstLo, int t) {
    #pragma unroll 4
    for (int i = 0; i < (16 * KDIM) / (NTHREADS * 4); ++i) {   // 16 iters of float4
        const int e = (i * NTHREADS + t) * 4;
        const float4 v = *(const float4*)(W + (size_t)16 * n * KDIM + e);
        float xs[4] = {v.x, v.y, v.z, v.w};
        v4bf hi, lo;
        #pragma unroll
        for (int j = 0; j < 4; ++j) {
            __bf16 h, l;
            split_bf16(xs[j], h, l);
            hi[j] = h; lo[j] = l;
        }
        *(v4bf*)(dstHi + e) = hi;
        *(v4bf*)(dstLo + e) = lo;
    }
}

// Pre-split W into bf16 hi/lo planes in workspace: ws[0..NTRIL*KDIM) = hi, then lo.
__global__ void w_split_kernel(const float* __restrict__ W, __bf16* __restrict__ ws) {
    int i = blockIdx.x * blockDim.x + threadIdx.x;
    if (i < NTRIL * KDIM) {
        __bf16 hi, lo;
        split_bf16(W[i], hi, lo);
        ws[i] = hi;
        ws[NTRIL * KDIM + i] = lo;
    }
}

__global__ __launch_bounds__(NTHREADS)
void advantage_kernel(const float* __restrict__ logits,
                      const float* __restrict__ best_action,
                      const float* __restrict__ actions,
                      const float* __restrict__ W,
                      const float* __restrict__ bias,
                      const __bf16* __restrict__ wsplit,
                      int useWs,
                      float* __restrict__ out) {
    extern __shared__ char smem[];
    __bf16* ldsAhi  = (__bf16*)(smem + OFF_AHI);
    __bf16* ldsAlo  = (__bf16*)(smem + OFF_ALO);
    float*  ldsDiff = (float*)(smem + OFF_DIFF);
    float*  ldsY    = (float*)(smem + OFF_Y);

    const int t       = threadIdx.x;
    const int rowBase = blockIdx.x * MT;
    const __bf16* wsHi = wsplit;
    const __bf16* wsLo = wsplit + (size_t)NTRIL * KDIM;

    // ---- Stage logits (64 rows x 512 f32) into LDS as split bf16 ----
    #pragma unroll 2
    for (int i = 0; i < MT; ++i) {
        const float4 v = *(const float4*)(logits + (size_t)(rowBase + i) * KDIM + t * 4);
        float xs[4] = {v.x, v.y, v.z, v.w};
        v4bf hi, lo;
        #pragma unroll
        for (int j = 0; j < 4; ++j) {
            __bf16 h, l;
            split_bf16(xs[j], h, l);
            hi[j] = h; lo[j] = l;
        }
        *(v4bf*)(ldsAhi + i * KDIM + t * 4) = hi;
        *(v4bf*)(ldsAlo + i * KDIM + t * 4) = lo;
    }
    // ---- diff = actions - best_action ; zero y ----
    #pragma unroll
    for (int i = 0; i < (MT * ADIM) / NTHREADS; ++i) {   // 16 iters
        int linear = i * NTHREADS + t;
        size_t g = (size_t)rowBase * ADIM + linear;
        ldsDiff[linear] = actions[g] - best_action[g];
        ldsY[linear]    = 0.0f;
    }

    // ---- Prologue: stage W tile 0 into buffer 0 ----
    {
        __bf16* d0hi = (__bf16*)(smem + OFF_W);
        __bf16* d0lo = (__bf16*)(smem + OFF_W + WTILE_BYTES);
        if (useWs) {
            async_w_tile(wsHi, wsLo, d0hi, d0lo, t);
            asm volatile("s_wait_asynccnt 0" ::: "memory");
        } else {
            convert_w_tile(W, 0, d0hi, d0lo, t);
        }
    }
    __syncthreads();

    const int lane = t & 31;
    const int wave = t >> 5;
    const int lr   = lane & 15;   // N index within tile / W row within tile
    const int hs   = lane >> 4;   // half-wave select
    const int aoff = hs * 16;     // A K-offset: contiguous k = 16*hs + s (matches B)
    const int boff = hs * 16;     // B K-offset: contiguous k = 16*hs + s (matches A)

    const __bf16* arowHi = ldsAhi + (wave * 16 + lr) * KDIM;
    const __bf16* arowLo = ldsAlo + (wave * 16 + lr) * KDIM;

    for (int n = 0; n < NTILES; ++n) {
        const int cur = n & 1, nxt = cur ^ 1;
        const bool havePrefetch = (n + 1 < NTILES);

        // runtime-computed double-buffer pointers (no pointer tables!)
        __bf16* curHi = (__bf16*)(smem + OFF_W + cur * WBUF_STRIDE);
        __bf16* curLo = (__bf16*)(smem + OFF_W + cur * WBUF_STRIDE + WTILE_BYTES);
        __bf16* nxtHi = (__bf16*)(smem + OFF_W + nxt * WBUF_STRIDE);
        __bf16* nxtLo = (__bf16*)(smem + OFF_W + nxt * WBUF_STRIDE + WTILE_BYTES);

        // ---- kick off staging of tile n+1 (overlaps WMMA below) ----
        if (havePrefetch) {
            if (useWs) {
                const size_t tb = (size_t)16 * (n + 1) * KDIM;
                async_w_tile(wsHi + tb, wsLo + tb, nxtHi, nxtLo, t);
            } else {
                convert_w_tile(W, n + 1, nxtHi, nxtLo, t);
            }
        }

        const __bf16* browHi = curHi + lr * KDIM;
        const __bf16* browLo = curLo + lr * KDIM;

        const int trilIdx = n * 16 + lr;
        const float bv = bias[trilIdx];
        v8f c = {bv, bv, bv, bv, bv, bv, bv, bv};

        // ---- K loop: 16 steps of K=32, 3 WMMAs each (bf16x3 split product) ----
        #pragma unroll 4
        for (int ks = 0; ks < KDIM / 32; ++ks) {
            const int kb = ks * 32;
            v8bf ah0 = *(const v8bf*)(arowHi + kb + aoff);
            v8bf ah1 = *(const v8bf*)(arowHi + kb + aoff + 8);
            v8bf al0 = *(const v8bf*)(arowLo + kb + aoff);
            v8bf al1 = *(const v8bf*)(arowLo + kb + aoff + 8);
            v16bf ahi = __builtin_shufflevector(ah0, ah1, 0,1,2,3,4,5,6,7,8,9,10,11,12,13,14,15);
            v16bf alo = __builtin_shufflevector(al0, al1, 0,1,2,3,4,5,6,7,8,9,10,11,12,13,14,15);
            v16bf bhi = *(const v16bf*)(browHi + kb + boff);
            v16bf blo = *(const v16bf*)(browLo + kb + boff);
            c = __builtin_amdgcn_wmma_f32_16x16x32_bf16(false, ahi, false, bhi, (short)0, c, false, false);
            c = __builtin_amdgcn_wmma_f32_16x16x32_bf16(false, ahi, false, blo, (short)0, c, false, false);
            c = __builtin_amdgcn_wmma_f32_16x16x32_bf16(false, alo, false, bhi, (short)0, c, false, false);
        }

        // ---- fused epilogue: scatter tril, square diag, accumulate y += L*diff ----
        const int r  = kTril.r[trilIdx];
        const int cc = kTril.c[trilIdx];
        #pragma unroll
        for (int g = 0; g < 8; ++g) {
            float val = c[g];
            if (r == cc) val *= val;                 // square diagonal entries
            int m = wave * 16 + hs * 8 + g;          // C-tile VGPR layout: M = g + 8*hs
            atomicAdd(&ldsY[m * ADIM + r], val * ldsDiff[m * ADIM + cc]);
        }

        if (useWs && havePrefetch)
            asm volatile("s_wait_asynccnt 0" ::: "memory");
        __syncthreads();   // publishes tile n+1, protects y ordering
    }

    // ---- advantage = -0.5 * ||y||^2 ----
    if (t < MT) {
        float s = 0.0f;
        #pragma unroll
        for (int rr = 0; rr < ADIM; ++rr) {
            float y = ldsY[t * ADIM + rr];
            s += y * y;
        }
        out[rowBase + t] = -0.5f * s;
    }
}

extern "C" void kernel_launch(void* const* d_in, const int* in_sizes, int n_in,
                              void* d_out, int out_size, void* d_ws, size_t ws_size,
                              hipStream_t stream) {
    const float* logits = (const float*)d_in[0];
    const float* best   = (const float*)d_in[1];
    const float* acts   = (const float*)d_in[2];
    const float* W      = (const float*)d_in[3];
    const float* bias   = (const float*)d_in[4];
    float* out = (float*)d_out;

    const size_t wsplitBytes = (size_t)2 * NTRIL * KDIM * 2;  // hi+lo bf16 planes
    const int useWs = (d_ws != nullptr && ws_size >= wsplitBytes) ? 1 : 0;
    __bf16* wsplit = (__bf16*)d_ws;
    if (useWs) {
        w_split_kernel<<<(NTRIL * KDIM + 255) / 256, 256, 0, stream>>>(W, wsplit);
    }

    (void)hipFuncSetAttribute((const void*)advantage_kernel,
                              hipFuncAttributeMaxDynamicSharedMemorySize, LDS_BYTES);
    advantage_kernel<<<BATCH / MT, NTHREADS, LDS_BYTES, stream>>>(
        logits, best, acts, W, bias, wsplit, useWs, out);
}
